// CrossChannelMultiheadAttention_25915832664175
// MI455X (gfx1250) — matrix operand order repaired and checked by reading, not compiled, not run on hardware
//
#include <hip/hip_runtime.h>
#include <hip/hip_bf16.h>

// ---------------------------------------------------------------------------
// CDNA5 (gfx1250, wave32) cross-channel MHA.
//   D=1024, H=8, dk=128, N=1024 tokens, bs2=16 (8 pairs x 2 channels).
//   Heads 0..5 attend the partner channel's K/V; heads 6..7 self.
// Pipeline:
//   0) one-time prep: x -> bf16; Wq/Wk/Wv/Wo -> bf16 transposed Wt[n][k]
//   1) QKV GEMMs (bf16 WMMA, f32 acc), Q pre-scaled, V stored transposed
//   2) flash attention with cross-channel K/V swap for heads 0..5
//   3) output projection -> f32
// All matmuls use v_wmma_f32_16x16x32_bf16; fragments fed by ds_load_b128.
// ---------------------------------------------------------------------------

typedef __attribute__((ext_vector_type(16))) __bf16 v16bf;
typedef __attribute__((ext_vector_type(8)))  __bf16 v8bf;
typedef __attribute__((ext_vector_type(8)))  float  v8f;
typedef __attribute__((ext_vector_type(4)))  float  v4f;

#define D_MODEL 1024
#define NTOK    1024
#define BS2     16
#define NHEADS  8
#define DK      128
#define NCROSS  6

// Build a 16x32 (bf16) WMMA operand from two contiguous 16B chunks.
__device__ __forceinline__ v16bf frag_pair(const __bf16* p0, const __bf16* p1) {
    v8bf lo = *(const v8bf*)p0;
    v8bf hi = *(const v8bf*)p1;
    v16bf r;
#pragma unroll
    for (int i = 0; i < 8; ++i) { r[i] = lo[i]; r[i + 8] = hi[i]; }
    return r;
}

// A-matrix 16x32 fragment from a row-major [16 x rs] bf16 tile.
// lane L: m = L%16; k = (L/16)*8 + 0..7  and  16+(L/16)*8 + 0..7
__device__ __forceinline__ v16bf load_afrag(const __bf16* base, int rs) {
    const int lane = threadIdx.x & 31;
    const __bf16* p = base + (lane & 15) * rs + (lane >> 4) * 8;
    return frag_pair(p, p + 16);
}

// B-matrix 32x16 fragment from an n-major (transposed) [16 x rs] bf16 tile:
// lane L holds n = L%16, k = (L/16)*8 + 0..7  and  16+(L/16)*8 + 0..7.
__device__ __forceinline__ v16bf load_bfrag(const __bf16* base, int rs) {
    const int lane = threadIdx.x & 31;
    const __bf16* p = base + (lane & 15) * rs + (lane >> 4) * 8;
    return frag_pair(p, p + 16);
}

__device__ __forceinline__ v8f wmma_bf16(v16bf a, v16bf b, v8f c) {
    return __builtin_amdgcn_wmma_f32_16x16x32_bf16(
        /*neg_a=*/false, a, /*neg_b=*/false, b,
        /*c_mod=*/(short)0, c, /*reuse_a=*/false, /*reuse_b=*/false);
}

// ---------------------------------------------------------------------------
// Prep 0a: x (f32) -> xb (bf16), 4 elements/thread.
// ---------------------------------------------------------------------------
__global__ __launch_bounds__(256) void convert_x_kernel(
    const float* __restrict__ x, __bf16* __restrict__ xb)
{
    const size_t i = ((size_t)blockIdx.x * 256 + threadIdx.x) * 4;
    v4f v = *(const v4f*)(x + i);
    __bf16* d = xb + i;
    d[0] = (__bf16)v.x; d[1] = (__bf16)v.y;
    d[2] = (__bf16)v.z; d[3] = (__bf16)v.w;
}

// ---------------------------------------------------------------------------
// Prep 0b: W[k][n] (f32) -> Wt[n][k] (bf16), LDS-tiled 64x64 transpose.
// blockIdx.z selects Wq/Wk/Wv/Wo.
// ---------------------------------------------------------------------------
__global__ __launch_bounds__(256) void transpose_w_kernel(
    const float* __restrict__ Wq, const float* __restrict__ Wk,
    const float* __restrict__ Wv, const float* __restrict__ Wo,
    __bf16* __restrict__ Wtq, __bf16* __restrict__ Wtk,
    __bf16* __restrict__ Wtv, __bf16* __restrict__ Wto)
{
    __shared__ __bf16 tile[64][72];                    // padded rows
    const int z = blockIdx.z;
    const float* W = (z == 0) ? Wq : (z == 1) ? Wk : (z == 2) ? Wv : Wo;
    __bf16*     Wt = (z == 0) ? Wtq : (z == 1) ? Wtk : (z == 2) ? Wtv : Wto;

    const int k0 = blockIdx.x * 64, n0 = blockIdx.y * 64;
    const int tx = threadIdx.x & 63, ty = threadIdx.x >> 6;
#pragma unroll
    for (int j = 0; j < 16; ++j) {
        const int r = ty + j * 4;                      // k within tile
        tile[r][tx] = (__bf16)W[(size_t)(k0 + r) * D_MODEL + n0 + tx];
    }
    __syncthreads();
#pragma unroll
    for (int j = 0; j < 16; ++j) {
        const int n = ty + j * 4;                      // n within tile
        Wt[(size_t)(n0 + n) * D_MODEL + k0 + tx] = tile[tx][n];
    }
}

// ---------------------------------------------------------------------------
// Kernel 1: fused QKV projection (all-bf16 operands).
// Block: 256 thr (8 waves). Block tile 64(M) x 256(N); wave tile 32 x 64.
// blockIdx.z selects Q/K/V. Outputs (bf16):
//   Q[b2,h,tok,d] pre-scaled by 1/sqrt(dk); K[b2,h,tok,d]; Vt[b2,h,d,tok].
// ---------------------------------------------------------------------------
__global__ __launch_bounds__(256) void gemm_qkv_kernel(
    const __bf16* __restrict__ xb,
    const __bf16* __restrict__ Wtq, const __bf16* __restrict__ Wtk,
    const __bf16* __restrict__ Wtv,
    const float* __restrict__ bq, const float* __restrict__ bk,
    const float* __restrict__ bv,
    __bf16* __restrict__ Qo, __bf16* __restrict__ Ko, __bf16* __restrict__ Vto)
{
    __shared__ __bf16 Ash[64 * 32];     // [m][k]
    __shared__ __bf16 Wsh[256 * 32];    // [n][k] (already transposed)

    const int mode = blockIdx.z;                       // 0=Q 1=K 2=V
    const __bf16* Wt   = (mode == 0) ? Wtq : (mode == 1) ? Wtk : Wtv;
    const float*  bias = (mode == 0) ? bq  : (mode == 1) ? bk  : bv;

    const int m0 = blockIdx.x * 64;
    const int n0 = blockIdx.y * 256;
    const int t  = threadIdx.x;
    const int wave = t >> 5, lane = t & 31;
    const int mw = wave & 1, nw = wave >> 1;           // 2 x 4 wave grid
    const int half = lane >> 4, nl = lane & 15;

    v8f acc[2][4] = {};

    for (int k0 = 0; k0 < D_MODEL; k0 += 32) {
        __syncthreads();
        {   // A tile 64x32 bf16: pure 16B copies.
            const int row = t >> 2, c = (t & 3) * 8;
            *(ulonglong2*)&Ash[row * 32 + c] =
                *(const ulonglong2*)&xb[(size_t)(m0 + row) * D_MODEL + k0 + c];
        }
        {   // W tile 256(n)x32(k) bf16: 64B per thread, coalesced rows.
            const __bf16* src = &Wt[(size_t)(n0 + t) * D_MODEL + k0];
            *(ulonglong4*)&Wsh[t * 32]      = *(const ulonglong4*)src;
            *(ulonglong4*)&Wsh[t * 32 + 16] = *(const ulonglong4*)(src + 16);
            __builtin_prefetch(src + 32, 0, 0);        // next K step
        }
        __syncthreads();

        v16bf a0 = load_afrag(&Ash[(mw * 32) * 32], 32);
        v16bf a1 = load_afrag(&Ash[(mw * 32 + 16) * 32], 32);
#pragma unroll
        for (int f = 0; f < 4; ++f) {
            v16bf b = load_bfrag(&Wsh[(nw * 64 + f * 16) * 32], 32);
            acc[0][f] = wmma_bf16(a0, b, acc[0][f]);
            acc[1][f] = wmma_bf16(a1, b, acc[1][f]);
        }
    }

    const float qscale = 0.08838834764831845f;         // 1/sqrt(128)
#pragma unroll
    for (int g = 0; g < 2; ++g) {
#pragma unroll
        for (int f = 0; f < 4; ++f) {
            const int ncol = n0 + nw * 64 + f * 16 + nl;
            const float bb = bias[ncol];
            const int hh = ncol >> 7, dd = ncol & 127; // head, dim
#pragma unroll
            for (int r = 0; r < 8; ++r) {
                const int m  = m0 + mw * 32 + g * 16 + r + half * 8;
                const int b2 = m >> 10, tok = m & 1023;
                const float v = acc[g][f][r] + bb;
                if (mode == 0) {
                    Qo[(((size_t)(b2 * NHEADS + hh) * NTOK + tok) << 7) + dd] =
                        (__bf16)(v * qscale);
                } else if (mode == 1) {
                    Ko[(((size_t)(b2 * NHEADS + hh) * NTOK + tok) << 7) + dd] =
                        (__bf16)v;
                } else {
                    Vto[(((size_t)(b2 * NHEADS + hh) * DK + dd) << 10) + tok] =
                        (__bf16)v;
                }
            }
        }
    }
}

// ---------------------------------------------------------------------------
// Kernel 2: flash attention per (b2, head). Block = 8 waves, 128 queries;
// each wave owns 16 queries. Streams keys in chunks of 32 through LDS.
// Heads 0..5 read K/V of partner channel (b2 ^ 1).
// Output O[b2, tok, h*dk + d] bf16 (A operand of the final projection).
// ---------------------------------------------------------------------------
__global__ __launch_bounds__(256) void attn_kernel(
    const __bf16* __restrict__ Q, const __bf16* __restrict__ K,
    const __bf16* __restrict__ Vt, __bf16* __restrict__ O)
{
    __shared__ __bf16 Ksh[32 * 128];        // [key][dim]
    __shared__ __bf16 Vtsh[128 * 32];       // [dim][key]
    __shared__ __bf16 Pbuf[8 * 16 * 32];    // per-wave P re-layout buffer

    const int b2 = blockIdx.z, h = blockIdx.y;
    const int kvb2 = (h < NCROSS) ? (b2 ^ 1) : b2;
    const int t = threadIdx.x, wave = t >> 5, lane = t & 31;
    const int half = lane >> 4, nl = lane & 15;
    const int q0 = blockIdx.x * 128 + wave * 16;

    const __bf16* Qb = Q  + ((size_t)(b2   * NHEADS + h) * NTOK << 7);
    const __bf16* Kb = K  + ((size_t)(kvb2 * NHEADS + h) * NTOK << 7);
    const __bf16* Vb = Vt + ((size_t)(kvb2 * NHEADS + h) * DK   << 10);

    // Preload this wave's Q tile as 4 A-fragments (16 x 128 total).
    v16bf qf[4];
#pragma unroll
    for (int ks = 0; ks < 4; ++ks) {
        const __bf16* p = Qb + (size_t)(q0 + nl) * DK + ks * 32 + half * 8;
        qf[ks] = frag_pair(p, p + 16);
    }

    float mi[8], li[8];
    v8f accO[8];
#pragma unroll
    for (int r = 0; r < 8; ++r) { mi[r] = -1e30f; li[r] = 0.0f; }
#pragma unroll
    for (int d = 0; d < 8; ++d) accO[d] = v8f{};

    __bf16* Pw = &Pbuf[wave * 16 * 32];
    const float LOG2E = 1.44269504088896f;

    for (int j0 = 0; j0 < NTOK; j0 += 32) {
        __syncthreads();
        // Stage K chunk (32 keys x 128 dims) and Vt chunk (128 dims x 32 keys).
        {
            const int row = t >> 3, c = (t & 7) * 16;  // 32B per thread
            *(ulonglong4*)&Ksh[row * 128 + c] =
                *(const ulonglong4*)&Kb[(size_t)(j0 + row) * DK + c];
        }
        {
            const int dim = t >> 1, c = (t & 1) * 16;
            *(ulonglong4*)&Vtsh[dim * 32 + c] =
                *(const ulonglong4*)&Vb[(size_t)dim * NTOK + j0 + c];
        }
        __syncthreads();

        // S = Q @ K^T for two 16-key tiles (scale pre-folded into Q).
        v8f s0 = v8f{}, s1 = v8f{};
#pragma unroll
        for (int ks = 0; ks < 4; ++ks) {
            v16bf kb0 = load_bfrag(&Ksh[0 * 128 + ks * 32], 128);
            v16bf kb1 = load_bfrag(&Ksh[16 * 128 + ks * 32], 128);
            s0 = wmma_bf16(qf[ks], kb0, s0);
            s1 = wmma_bf16(qf[ks], kb1, s1);
        }

        // Online softmax. Row r+half*8 lives across the 16 lanes of a half.
        float alpha[8];
#pragma unroll
        for (int r = 0; r < 8; ++r) {
            float mx = fmaxf(s0[r], s1[r]);
            mx = fmaxf(mx, __shfl_xor(mx, 1));
            mx = fmaxf(mx, __shfl_xor(mx, 2));
            mx = fmaxf(mx, __shfl_xor(mx, 4));
            mx = fmaxf(mx, __shfl_xor(mx, 8));
            const float mn = fmaxf(mi[r], mx);
            alpha[r] = __builtin_exp2f((mi[r] - mn) * LOG2E);
            const float p0 = __builtin_exp2f((s0[r] - mn) * LOG2E);
            const float p1 = __builtin_exp2f((s1[r] - mn) * LOG2E);
            float rs = p0 + p1;
            rs += __shfl_xor(rs, 1);
            rs += __shfl_xor(rs, 2);
            rs += __shfl_xor(rs, 4);
            rs += __shfl_xor(rs, 8);
            li[r] = li[r] * alpha[r] + rs;
            mi[r] = mn;
            const int row = r + half * 8;
            Pw[row * 32 + nl]      = (__bf16)p0;       // C-layout -> LDS
            Pw[row * 32 + 16 + nl] = (__bf16)p1;
        }
#pragma unroll
        for (int d = 0; d < 8; ++d)
#pragma unroll
            for (int r = 0; r < 8; ++r) accO[d][r] = accO[d][r] * alpha[r];

        // Reload P in A-fragment layout (16 x 32), then O += P @ V.
        const __bf16* pp = Pw + nl * 32 + half * 8;
        v16bf pf = frag_pair(pp, pp + 16);
#pragma unroll
        for (int d = 0; d < 8; ++d) {
            v16bf vb = load_bfrag(&Vtsh[(d * 16) * 32], 32);
            accO[d] = wmma_bf16(pf, vb, accO[d]);
        }
    }

    // Normalize and store O[b2, tok, h*128 + d] as bf16.
#pragma unroll
    for (int d = 0; d < 8; ++d) {
        const int dcol = h * DK + d * 16 + nl;
#pragma unroll
        for (int r = 0; r < 8; ++r) {
            const int tok = q0 + r + half * 8;
            O[((size_t)(b2 * NTOK + tok) << 10) + dcol] =
                (__bf16)(accO[d][r] / li[r]);
        }
    }
}

// ---------------------------------------------------------------------------
// Kernel 3: output projection. out[m,n] = O[m,:] @ Wo[:,n] + bo[n]  (f32 out)
// Same tiling as QKV GEMM; Wto is pre-transposed bf16.
// ---------------------------------------------------------------------------
__global__ __launch_bounds__(256) void gemm_out_kernel(
    const __bf16* __restrict__ A, const __bf16* __restrict__ Wto,
    const float* __restrict__ bo, float* __restrict__ out)
{
    __shared__ __bf16 Ash[64 * 32];
    __shared__ __bf16 Wsh[256 * 32];

    const int m0 = blockIdx.x * 64, n0 = blockIdx.y * 256;
    const int t = threadIdx.x, wave = t >> 5, lane = t & 31;
    const int mw = wave & 1, nw = wave >> 1;
    const int half = lane >> 4, nl = lane & 15;

    v8f acc[2][4] = {};

    for (int k0 = 0; k0 < D_MODEL; k0 += 32) {
        __syncthreads();
        {
            const int row = t >> 2, c = (t & 3) * 8;
            *(ulonglong2*)&Ash[row * 32 + c] =
                *(const ulonglong2*)&A[(size_t)(m0 + row) * D_MODEL + k0 + c];
        }
        {
            const __bf16* src = &Wto[(size_t)(n0 + t) * D_MODEL + k0];
            *(ulonglong4*)&Wsh[t * 32]      = *(const ulonglong4*)src;
            *(ulonglong4*)&Wsh[t * 32 + 16] = *(const ulonglong4*)(src + 16);
            __builtin_prefetch(src + 32, 0, 0);
        }
        __syncthreads();

        v16bf a0 = load_afrag(&Ash[(mw * 32) * 32], 32);
        v16bf a1 = load_afrag(&Ash[(mw * 32 + 16) * 32], 32);
#pragma unroll
        for (int f = 0; f < 4; ++f) {
            v16bf b = load_bfrag(&Wsh[(nw * 64 + f * 16) * 32], 32);
            acc[0][f] = wmma_bf16(a0, b, acc[0][f]);
            acc[1][f] = wmma_bf16(a1, b, acc[1][f]);
        }
    }

#pragma unroll
    for (int g = 0; g < 2; ++g) {
#pragma unroll
        for (int f = 0; f < 4; ++f) {
            const int ncol = n0 + nw * 64 + f * 16 + nl;
            const float bb = bo[ncol];
#pragma unroll
            for (int r = 0; r < 8; ++r) {
                const int m = m0 + mw * 32 + g * 16 + r + half * 8;
                out[(size_t)m * D_MODEL + ncol] = acc[g][f][r] + bb;
            }
        }
    }
}

// ---------------------------------------------------------------------------
extern "C" void kernel_launch(void* const* d_in, const int* in_sizes, int n_in,
                              void* d_out, int out_size, void* d_ws, size_t ws_size,
                              hipStream_t stream) {
    const float* x  = (const float*)d_in[0];
    const float* Wq = (const float*)d_in[1];
    const float* bq = (const float*)d_in[2];
    const float* Wk = (const float*)d_in[3];
    const float* bk = (const float*)d_in[4];
    const float* Wv = (const float*)d_in[5];
    const float* bv = (const float*)d_in[6];
    const float* Wo = (const float*)d_in[7];
    const float* bo = (const float*)d_in[8];
    float* out = (float*)d_out;

    const size_t xelems = (size_t)BS2 * NTOK * D_MODEL;      // 16,777,216
    const size_t welems = (size_t)D_MODEL * D_MODEL;         //  1,048,576
    __bf16* xb  = (__bf16*)d_ws;
    __bf16* Wtq = xb  + xelems;
    __bf16* Wtk = Wtq + welems;
    __bf16* Wtv = Wtk + welems;
    __bf16* Wto = Wtv + welems;
    __bf16* Qw  = Wto + welems;
    __bf16* Kw  = Qw  + xelems;
    __bf16* Vtw = Kw  + xelems;
    __bf16* Ow  = Vtw + xelems;                              // [bs2*N][D]

    // 0) One-time precision/layout prep.
    convert_x_kernel<<<dim3(16384), 256, 0, stream>>>(x, xb);
    transpose_w_kernel<<<dim3(16, 16, 4), 256, 0, stream>>>(
        Wq, Wk, Wv, Wo, Wtq, Wtk, Wtv, Wto);
    // 1) QKV projections (z selects Q/K/V).
    gemm_qkv_kernel<<<dim3(256, 4, 3), 256, 0, stream>>>(
        xb, Wtq, Wtk, Wtv, bq, bk, bv, Qw, Kw, Vtw);
    // 2) Cross/self flash attention.
    attn_kernel<<<dim3(8, NHEADS, BS2), 256, 0, stream>>>(Qw, Kw, Vtw, Ow);
    // 3) Output projection.
    gemm_out_kernel<<<dim3(256, 4, 1), 256, 0, stream>>>(Ow, Wto, bo, out);
}
